// Model_44444321579745
// MI455X (gfx1250) — matrix operand order repaired and checked by reading, not compiled, not run on hardware
//
#include <hip/hip_runtime.h>

typedef __bf16 bf16_t;
typedef __attribute__((ext_vector_type(16))) __bf16 v16bf;
typedef __attribute__((ext_vector_type(8)))  __bf16 v8bf;
typedef __attribute__((ext_vector_type(8)))  float  v8f;
typedef __attribute__((ext_vector_type(4)))  int    v4i;

#define AS_GLOBAL __attribute__((address_space(1)))
#define AS_LDS    __attribute__((address_space(3)))

#define H 128
#define LN_EPS 1e-5f

// Async global->LDS DMA path (ASYNCcnt), if this toolchain exposes the builtins.
#if __has_builtin(__builtin_amdgcn_global_load_async_to_lds_b128) && \
    __has_builtin(__builtin_amdgcn_s_wait_asynccnt)
#define USE_ASYNC_TILE 1
#else
#define USE_ASYNC_TILE 0
#endif

// ---------------------------------------------------------------------------
// WMMA fragment loaders (layouts per CDNA5 ISA 7.12.2, wave32)
// ---------------------------------------------------------------------------

// A fragment: 16x32 bf16 tile in LDS (row stride H halves, base 32B-aligned).
// lanes 0-15: row=lane, halves 0..7 -> K kbase+0..7,  halves 8..15 -> K kbase+16..23
// lanes 16-31: row=lane-16, halves 0..7 -> K kbase+8..15, halves 8..15 -> K kbase+24..31
__device__ __forceinline__ v16bf load_a_frag(const bf16_t A[][H], int kbase, int lane) {
    const int row = lane & 15;
    const int c0  = (lane >> 4) * 8;
    v8bf lo = *(const v8bf*)&A[row][kbase + c0];
    v8bf hi = *(const v8bf*)&A[row][kbase + 16 + c0];
    return __builtin_shufflevector(lo, hi, 0, 1, 2, 3, 4, 5, 6, 7,
                                           8, 9, 10, 11, 12, 13, 14, 15);
}

// B fragment from fragment-packed weights: one aligned 32-byte load per lane.
//   Wp[(((ktile*8 + ntile)*32) + lane)*16 + i] = W[(ktile*32 + (lane>>4)*8 + i + (i>>3)*8)*128
//                                                  + ntile*16 + (lane&15)]
__device__ __forceinline__ v16bf load_b_packed(const bf16_t* __restrict__ Wp,
                                               int ktile, int ntile, int lane) {
    return *(const v16bf*)(Wp + (((ktile * 8 + ntile) * 32 + lane) * 16));
}

// 16x16 output tile of A(16x128) @ W(128 x [ntile*16 .. ntile*16+16))
__device__ __forceinline__ v8f wmma_gemm128(const bf16_t A[][H], const bf16_t* __restrict__ Wp,
                                            int ntile, int lane, v8f acc) {
#pragma unroll
    for (int kt = 0; kt < 4; ++kt) {
        v16bf a = load_a_frag(A, kt * 32, lane);
        v16bf b = load_b_packed(Wp, kt, ntile, lane);
        acc = __builtin_amdgcn_wmma_f32_16x16x32_bf16(false, a, false, b,
                                                      (short)0, acc, false, false);
    }
    return acc;
}

// Row-wise LayerNorm stats over u[16][128]: 8 waves x 2 rows, lane-parallel reduce.
__device__ __forceinline__ void ln_stats(const float u[][H], float* mu, float* rv, int tid) {
    const int wv = tid >> 5, lane = tid & 31;
#pragma unroll
    for (int rr = 0; rr < 2; ++rr) {
        const int r = wv * 2 + rr;
        float s = 0.f, s2 = 0.f;
#pragma unroll
        for (int c = 0; c < 4; ++c) {
            float v = u[r][lane + 32 * c];
            s += v; s2 += v * v;
        }
#pragma unroll
        for (int m = 16; m; m >>= 1) {
            s  += __shfl_xor(s, m, 32);
            s2 += __shfl_xor(s2, m, 32);
        }
        if (lane == 0) {
            float mean = s * (1.f / 128.f);
            mu[r] = mean;
            rv[r] = rsqrtf(s2 * (1.f / 128.f) - mean * mean + LN_EPS);
        }
    }
}

// ---------------------------------------------------------------------------
// f32 weights -> bf16, swizzled into WMMA B-fragment order (coalesced writes)
// ---------------------------------------------------------------------------
__global__ void cvt_swizzle_kernel(const float* __restrict__ src, bf16_t* __restrict__ dst,
                                   int n) {
    int d = blockIdx.x * blockDim.x + threadIdx.x;
    if (d >= n) return;
    int i     = d & 15;
    int lane  = (d >> 4) & 31;
    int ntile = (d >> 9) & 7;
    int ktile = d >> 12;
    int k = ktile * 32 + ((lane >> 4) << 3) + i + ((i >> 3) << 3);
    int c = ntile * 16 + (lane & 15);
    dst[d] = (bf16_t)src[k * H + c];
}

// ---------------------------------------------------------------------------
// K=10 neighbor attention: one wave per (b,n) row.  agg[b,n,:] out (f32)
// ---------------------------------------------------------------------------
__global__ void __launch_bounds__(256)
attn_kernel(const float* __restrict__ x, const int* __restrict__ nidx,
            float* __restrict__ agg) {
    const int wave = (blockIdx.x * blockDim.x + threadIdx.x) >> 5; // [0,3200)
    const int lane = threadIdx.x & 31;
    const int b = wave / 100;

    float xv[4];
#pragma unroll
    for (int c = 0; c < 4; ++c) xv[c] = x[wave * H + lane + 32 * c];

    float nbv[10][4];
    float score[10];
#pragma unroll
    for (int k = 0; k < 10; ++k) {
        const int j = nidx[wave * 10 + k];
        const float* nr = x + (b * 100 + j) * H;
        float s = 0.f;
#pragma unroll
        for (int c = 0; c < 4; ++c) {
            nbv[k][c] = nr[lane + 32 * c];
            s += xv[c] * nbv[k][c];
        }
#pragma unroll
        for (int m = 16; m; m >>= 1) s += __shfl_xor(s, m, 32);
        score[k] = s * 0.08838834764831845f; // 1/sqrt(128)
    }
    float mx = score[0];
#pragma unroll
    for (int k = 1; k < 10; ++k) mx = fmaxf(mx, score[k]);
    float den = 0.f;
#pragma unroll
    for (int k = 0; k < 10; ++k) { score[k] = __expf(score[k] - mx); den += score[k]; }
    const float inv = 1.f / den;
    float out[4] = {0.f, 0.f, 0.f, 0.f};
#pragma unroll
    for (int k = 0; k < 10; ++k)
#pragma unroll
        for (int c = 0; c < 4; ++c) out[c] += score[k] * nbv[k][c];
#pragma unroll
    for (int c = 0; c < 4; ++c) agg[wave * H + lane + 32 * c] = out[c] * inv;
}

// ---------------------------------------------------------------------------
// Fused node block: h_node (out), plus w2x = x@W2e+b2, w3x = x@W3e+b3 (scratch)
// ---------------------------------------------------------------------------
__global__ void __launch_bounds__(256)
node_kernel(const float* __restrict__ x, const float* __restrict__ agg,
            const bf16_t* __restrict__ Wn,  const float* __restrict__ Wn_b,
            const bf16_t* __restrict__ Vni, const float* __restrict__ Vni_b,
            const bf16_t* __restrict__ Vn,  const float* __restrict__ Vn_b,
            const bf16_t* __restrict__ W2e, const float* __restrict__ W2e_b,
            const bf16_t* __restrict__ W3e, const float* __restrict__ W3e_b,
            const float* __restrict__ ln1g, const float* __restrict__ ln1b,
            const float* __restrict__ ln2g, const float* __restrict__ ln2b,
            float* __restrict__ w2x, float* __restrict__ w3x,
            float* __restrict__ h_node) {
    __shared__ alignas(32) float  xs[16][H];
    __shared__ alignas(32) bf16_t xb[16][H];
    __shared__ alignas(32) bf16_t ab[16][H];
    __shared__ alignas(32) float  u [16][H];
    __shared__ alignas(32) float  hf[16][H];
    __shared__ alignas(32) bf16_t hb[16][H];
    __shared__ alignas(32) bf16_t vb[16][H];
    __shared__ float  mu[16], rv[16];

    const int tid   = threadIdx.x;
    const int lane  = tid & 31;
    const int wv    = tid >> 5;
    const int rbase = blockIdx.x * 16;
    const int hi    = lane >> 4;
    const int col   = wv * 16 + (lane & 15);

#pragma unroll
    for (int i = 0; i < 2; ++i) {
        int f4 = i * 256 + tid, r = f4 >> 5, c = (f4 & 31) << 2;
        float4 vx = *(const float4*)&x[(rbase + r) * H + c];
        float4 va = *(const float4*)&agg[(rbase + r) * H + c];
        xs[r][c] = vx.x; xs[r][c + 1] = vx.y; xs[r][c + 2] = vx.z; xs[r][c + 3] = vx.w;
        xb[r][c] = (bf16_t)vx.x; xb[r][c + 1] = (bf16_t)vx.y;
        xb[r][c + 2] = (bf16_t)vx.z; xb[r][c + 3] = (bf16_t)vx.w;
        ab[r][c] = (bf16_t)va.x; ab[r][c + 1] = (bf16_t)va.y;
        ab[r][c + 2] = (bf16_t)va.z; ab[r][c + 3] = (bf16_t)va.w;
    }
    __syncthreads();

    { // w2x = x@W2e + b ; w3x = x@W3e + b   (feeds the edge kernel)
        v8f a2 = {}; a2 = wmma_gemm128(xb, W2e, wv, lane, a2);
        float bias2 = W2e_b[col];
#pragma unroll
        for (int r = 0; r < 8; ++r)
            w2x[(rbase + r + 8 * hi) * H + col] = a2[r] + bias2;
        v8f a3 = {}; a3 = wmma_gemm128(xb, W3e, wv, lane, a3);
        float bias3 = W3e_b[col];
#pragma unroll
        for (int r = 0; r < 8; ++r)
            w3x[(rbase + r + 8 * hi) * H + col] = a3[r] + bias3;
    }
    { // vni = x@Vni + b
        v8f a1 = {}; a1 = wmma_gemm128(xb, Vni, wv, lane, a1);
        float bias = Vni_b[col];
#pragma unroll
        for (int r = 0; r < 8; ++r)
            vb[r + 8 * hi][col] = (bf16_t)(a1[r] + bias);
    }
    { // u = x + relu(agg@Wn + b)
        v8f a0 = {}; a0 = wmma_gemm128(ab, Wn, wv, lane, a0);
        float bias = Wn_b[col];
#pragma unroll
        for (int r = 0; r < 8; ++r) {
            int m = r + 8 * hi;
            u[m][col] = xs[m][col] + fmaxf(a0[r] + bias, 0.f);
        }
    }
    __syncthreads();
    ln_stats(u, mu, rv, tid);
    __syncthreads();
#pragma unroll
    for (int i = 0; i < 2; ++i) {
        int f4 = i * 256 + tid, r = f4 >> 5, c = (f4 & 31) << 2;
        float4 g = *(const float4*)&ln1g[c];
        float4 bb = *(const float4*)&ln1b[c];
        float m_ = mu[r], rs = rv[r];
        float v0 = (u[r][c + 0] - m_) * rs * g.x + bb.x;
        float v1 = (u[r][c + 1] - m_) * rs * g.y + bb.y;
        float v2 = (u[r][c + 2] - m_) * rs * g.z + bb.z;
        float v3 = (u[r][c + 3] - m_) * rs * g.w + bb.w;
        *(float4*)&hf[r][c] = make_float4(v0, v1, v2, v3);
        hb[r][c] = (bf16_t)v0; hb[r][c + 1] = (bf16_t)v1;
        hb[r][c + 2] = (bf16_t)v2; hb[r][c + 3] = (bf16_t)v3;
    }
    __syncthreads();
    { // u = h_nb + relu([vni|h_nb] @ Vn + b)  (256-K GEMM as two 128-K passes)
        v8f a4 = {};
        a4 = wmma_gemm128(vb, Vn,         wv, lane, a4);
        a4 = wmma_gemm128(hb, Vn + H * H, wv, lane, a4);
        float bias = Vn_b[col];
#pragma unroll
        for (int r = 0; r < 8; ++r) {
            int m = r + 8 * hi;
            u[m][col] = hf[m][col] + fmaxf(a4[r] + bias, 0.f);
        }
    }
    __syncthreads();
    ln_stats(u, mu, rv, tid);
    __syncthreads();
#pragma unroll
    for (int i = 0; i < 2; ++i) {
        int f4 = i * 256 + tid, r = f4 >> 5, c = (f4 & 31) << 2;
        float4 g = *(const float4*)&ln2g[c];
        float4 bb = *(const float4*)&ln2b[c];
        float m_ = mu[r], rs = rv[r];
        float4 o;
        o.x = (u[r][c + 0] - m_) * rs * g.x + bb.x;
        o.y = (u[r][c + 1] - m_) * rs * g.y + bb.y;
        o.z = (u[r][c + 2] - m_) * rs * g.z + bb.z;
        o.w = (u[r][c + 3] - m_) * rs * g.w + bb.w;
        *(float4*)&h_node[(rbase + r) * H + c] = o;
    }
}

// ---------------------------------------------------------------------------
// Fused edge block over 320,000 rows of e: 4 WMMA GEMMs, 2 LayerNorms;
// e tile arrives via async global->LDS DMA (ASYNCcnt) when available.
// ---------------------------------------------------------------------------
__global__ void __launch_bounds__(256)
edge_kernel(const float* __restrict__ e,
            const bf16_t* __restrict__ W1e, const float* __restrict__ W1e_b,
            const bf16_t* __restrict__ We,  const float* __restrict__ We_b,
            const bf16_t* __restrict__ Vei, const float* __restrict__ Vei_b,
            const bf16_t* __restrict__ Ve,  const float* __restrict__ Ve_b,
            const float* __restrict__ ln1g, const float* __restrict__ ln1b,
            const float* __restrict__ ln2g, const float* __restrict__ ln2b,
            const float* __restrict__ w2x,  const float* __restrict__ w3x,
            float* __restrict__ h_edge) {
    __shared__ alignas(32) float  ef[16][H];
    __shared__ alignas(32) bf16_t eb[16][H];
    __shared__ alignas(32) bf16_t pb[16][H];
    __shared__ alignas(32) float  u [16][H];
    __shared__ alignas(32) float  hf[16][H];
    __shared__ alignas(32) bf16_t hb[16][H];
    __shared__ alignas(32) bf16_t vb[16][H];
    __shared__ float  mu[16], rv[16];

    const int tid   = threadIdx.x;
    const int lane  = tid & 31;
    const int wv    = tid >> 5;
    const int rbase = blockIdx.x * 16;
    const int hi    = lane >> 4;
    const int col   = wv * 16 + (lane & 15);

    // speculative prefetch of the next block's e tile (global_prefetch_b8)
    if (blockIdx.x + 1 < gridDim.x)
        __builtin_prefetch(&e[(rbase + 16) * H + (tid << 3)], 0, 0);

#if USE_ASYNC_TILE
    { // DMA the 8KB f32 tile into LDS: 2 x b128 per thread, tracked by ASYNCcnt
        const char* gsrc = (const char*)&e[rbase * H];
        char*       ldst = (char*)&ef[0][0];
#pragma unroll
        for (int i = 0; i < 2; ++i) {
            int off = (i * 256 + tid) << 4;
            __builtin_amdgcn_global_load_async_to_lds_b128(
                (AS_GLOBAL v4i*)(unsigned long long)(gsrc + off),
                (AS_LDS v4i*)(unsigned)(unsigned long long)(ldst + off), 0, 0);
        }
        __builtin_amdgcn_s_wait_asynccnt(0);
        __syncthreads();
        // convert to bf16 from LDS
#pragma unroll
        for (int i = 0; i < 2; ++i) {
            int f4 = i * 256 + tid, r = f4 >> 5, c = (f4 & 31) << 2;
            float4 v = *(const float4*)&ef[r][c];
            eb[r][c] = (bf16_t)v.x; eb[r][c + 1] = (bf16_t)v.y;
            eb[r][c + 2] = (bf16_t)v.z; eb[r][c + 3] = (bf16_t)v.w;
        }
    }
#else
#pragma unroll
    for (int i = 0; i < 2; ++i) {
        int f4 = i * 256 + tid, r = f4 >> 5, c = (f4 & 31) << 2;
        float4 v = *(const float4*)&e[(rbase + r) * H + c];
        ef[r][c] = v.x; ef[r][c + 1] = v.y; ef[r][c + 2] = v.z; ef[r][c + 3] = v.w;
        eb[r][c] = (bf16_t)v.x; eb[r][c + 1] = (bf16_t)v.y;
        eb[r][c + 2] = (bf16_t)v.z; eb[r][c + 3] = (bf16_t)v.w;
    }
#endif
    __syncthreads();

    { // pre = e@W1e + b1 + w2x[b,i] + w3x[b,j]  (no relu)
        v8f a0 = {}; a0 = wmma_gemm128(eb, W1e, wv, lane, a0);
        float bias = W1e_b[col];
#pragma unroll
        for (int r = 0; r < 8; ++r) {
            int m  = r + 8 * hi;
            int rr = rbase + m;              // global edge row = (b*100 + i)*100 + j
            int b  = rr / 10000;
            int ij = rr - b * 10000;
            int ii = ij / 100;
            int jj = ij - ii * 100;
            float t = a0[r] + bias
                    + w2x[(b * 100 + ii) * H + col]
                    + w3x[(b * 100 + jj) * H + col];
            pb[m][col] = (bf16_t)t;
        }
    }
    { // vei = e@Vei + b
        v8f a1 = {}; a1 = wmma_gemm128(eb, Vei, wv, lane, a1);
        float bias = Vei_b[col];
#pragma unroll
        for (int r = 0; r < 8; ++r)
            vb[r + 8 * hi][col] = (bf16_t)(a1[r] + bias);
    }
    __syncthreads();
    { // u = e + relu(pre@We + b)
        v8f a2 = {}; a2 = wmma_gemm128(pb, We, wv, lane, a2);
        float bias = We_b[col];
#pragma unroll
        for (int r = 0; r < 8; ++r) {
            int m = r + 8 * hi;
            u[m][col] = ef[m][col] + fmaxf(a2[r] + bias, 0.f);
        }
    }
    __syncthreads();
    ln_stats(u, mu, rv, tid);
    __syncthreads();
#pragma unroll
    for (int i = 0; i < 2; ++i) {
        int f4 = i * 256 + tid, r = f4 >> 5, c = (f4 & 31) << 2;
        float4 g = *(const float4*)&ln1g[c];
        float4 bb = *(const float4*)&ln1b[c];
        float m_ = mu[r], rs = rv[r];
        float v0 = (u[r][c + 0] - m_) * rs * g.x + bb.x;
        float v1 = (u[r][c + 1] - m_) * rs * g.y + bb.y;
        float v2 = (u[r][c + 2] - m_) * rs * g.z + bb.z;
        float v3 = (u[r][c + 3] - m_) * rs * g.w + bb.w;
        *(float4*)&hf[r][c] = make_float4(v0, v1, v2, v3);
        hb[r][c] = (bf16_t)v0; hb[r][c + 1] = (bf16_t)v1;
        hb[r][c + 2] = (bf16_t)v2; hb[r][c + 3] = (bf16_t)v3;
    }
    __syncthreads();
    { // u = h_nb + relu([vei|h_nb] @ Ve + b)
        v8f a3 = {};
        a3 = wmma_gemm128(vb, Ve,         wv, lane, a3);
        a3 = wmma_gemm128(hb, Ve + H * H, wv, lane, a3);
        float bias = Ve_b[col];
#pragma unroll
        for (int r = 0; r < 8; ++r) {
            int m = r + 8 * hi;
            u[m][col] = hf[m][col] + fmaxf(a3[r] + bias, 0.f);
        }
    }
    __syncthreads();
    ln_stats(u, mu, rv, tid);
    __syncthreads();
#pragma unroll
    for (int i = 0; i < 2; ++i) {
        int f4 = i * 256 + tid, r = f4 >> 5, c = (f4 & 31) << 2;
        float4 g = *(const float4*)&ln2g[c];
        float4 bb = *(const float4*)&ln2b[c];
        float m_ = mu[r], rs = rv[r];
        float4 o;
        o.x = (u[r][c + 0] - m_) * rs * g.x + bb.x;
        o.y = (u[r][c + 1] - m_) * rs * g.y + bb.y;
        o.z = (u[r][c + 2] - m_) * rs * g.z + bb.z;
        o.w = (u[r][c + 3] - m_) * rs * g.w + bb.w;
        *(float4*)&h_edge[(rbase + r) * H + c] = o;
    }
}

// ---------------------------------------------------------------------------
extern "C" void kernel_launch(void* const* d_in, const int* in_sizes, int n_in,
                              void* d_out, int out_size, void* d_ws, size_t ws_size,
                              hipStream_t stream) {
    const float* x     = (const float*)d_in[0];
    const float* e     = (const float*)d_in[1];
    const int*   nidx  = (const int*)  d_in[2];
    const float* Wn_w  = (const float*)d_in[3];  const float* Wn_b  = (const float*)d_in[4];
    const float* Vni_w = (const float*)d_in[5];  const float* Vni_b = (const float*)d_in[6];
    const float* Vn_w  = (const float*)d_in[7];  const float* Vn_b  = (const float*)d_in[8];
    const float* ln1n_g = (const float*)d_in[9];  const float* ln1n_b = (const float*)d_in[10];
    const float* ln2n_g = (const float*)d_in[11]; const float* ln2n_b = (const float*)d_in[12];
    const float* We_w  = (const float*)d_in[13]; const float* We_b  = (const float*)d_in[14];
    const float* Vei_w = (const float*)d_in[15]; const float* Vei_b = (const float*)d_in[16];
    const float* Ve_w  = (const float*)d_in[17]; const float* Ve_b  = (const float*)d_in[18];
    const float* W1e_w = (const float*)d_in[19]; const float* W1e_b = (const float*)d_in[20];
    const float* W2e_w = (const float*)d_in[21]; const float* W2e_b = (const float*)d_in[22];
    const float* W3e_w = (const float*)d_in[23]; const float* W3e_b = (const float*)d_in[24];
    const float* ln1e_g = (const float*)d_in[25]; const float* ln1e_b = (const float*)d_in[26];
    const float* ln2e_g = (const float*)d_in[27]; const float* ln2e_b = (const float*)d_in[28];

    char* ws = (char*)d_ws;
    bf16_t* wbf    = (bf16_t*)ws;            // fragment-packed bf16 weights
    bf16_t* Wn_bf  = wbf + 0;       // 128x128
    bf16_t* Vni_bf = wbf + 16384;   // 128x128
    bf16_t* Vn_bf  = wbf + 32768;   // 256x128
    bf16_t* We_bf  = wbf + 65536;   // 128x128
    bf16_t* Vei_bf = wbf + 81920;   // 128x128
    bf16_t* Ve_bf  = wbf + 98304;   // 256x128
    bf16_t* W1e_bf = wbf + 131072;  // 128x128
    bf16_t* W2e_bf = wbf + 147456;  // 128x128
    bf16_t* W3e_bf = wbf + 163840;  // 128x128
    float* agg = (float*)(ws + (512u  << 10)); // 0x080000, 1.6 MB
    float* w2x = (float*)(ws + (2304u << 10)); // 0x240000, 1.6 MB
    float* w3x = (float*)(ws + (4096u << 10)); // 0x400000, 1.6 MB

    float* h_node = (float*)d_out;
    float* h_edge = (float*)d_out + 32 * 100 * H; // 409,600 floats in

    // 1) weights -> bf16, swizzled to WMMA B-fragment order
    cvt_swizzle_kernel<<<64, 256, 0, stream>>>(Wn_w,  Wn_bf,  16384);
    cvt_swizzle_kernel<<<64, 256, 0, stream>>>(Vni_w, Vni_bf, 16384);
    cvt_swizzle_kernel<<<128,256, 0, stream>>>(Vn_w,  Vn_bf,  32768);
    cvt_swizzle_kernel<<<64, 256, 0, stream>>>(We_w,  We_bf,  16384);
    cvt_swizzle_kernel<<<64, 256, 0, stream>>>(Vei_w, Vei_bf, 16384);
    cvt_swizzle_kernel<<<128,256, 0, stream>>>(Ve_w,  Ve_bf,  32768);
    cvt_swizzle_kernel<<<64, 256, 0, stream>>>(W1e_w, W1e_bf, 16384);
    cvt_swizzle_kernel<<<64, 256, 0, stream>>>(W2e_w, W2e_bf, 16384);
    cvt_swizzle_kernel<<<64, 256, 0, stream>>>(W3e_w, W3e_bf, 16384);

    // 2) neighbor attention -> agg   (3200 waves, 8 waves/block)
    attn_kernel<<<400, 256, 0, stream>>>(x, nidx, agg);

    // 3) fused node block (+ w2x/w3x for the edge path): 3200 rows / 16
    node_kernel<<<200, 256, 0, stream>>>(x, agg,
                                         Wn_bf, Wn_b, Vni_bf, Vni_b, Vn_bf, Vn_b,
                                         W2e_bf, W2e_b, W3e_bf, W3e_b,
                                         ln1n_g, ln1n_b, ln2n_g, ln2n_b,
                                         w2x, w3x, h_node);

    // 4) fused edge block: 320,000 rows / 16
    edge_kernel<<<20000, 256, 0, stream>>>(e,
                                           W1e_bf, W1e_b, We_bf, We_b,
                                           Vei_bf, Vei_b, Ve_bf, Ve_b,
                                           ln1e_g, ln1e_b, ln2e_g, ln2e_b,
                                           w2x, w3x, h_edge);

    (void)in_sizes; (void)n_in; (void)out_size; (void)ws_size;
}
